// CrossAttention_24086176596014
// MI455X (gfx1250) — matrix operand order repaired and checked by reading, not compiled, not run on hardware
//
#include <hip/hip_runtime.h>
#include <hip/hip_bf16.h>

typedef __attribute__((ext_vector_type(16))) __bf16        v16bf;
typedef __attribute__((ext_vector_type(8)))  float         v8f;
typedef __attribute__((ext_vector_type(4)))  unsigned int  u32x4;

#define HEADS    16
#define DIM_HEAD 64
#define DMODEL   1024
#define NQ       2048
#define NK       1024
#define BATCH    4
#define ATTN_SCALE 0.125f   // 64^-0.5, folded into Q projection

// ---------- helpers ----------

// Native f32 -> bf16 (RTNE); let the backend pick v_cvt_*_bf16 hardware ops.
__device__ __forceinline__ unsigned short f2bf(float f) {
  __bf16 h = (__bf16)f;
  return __builtin_bit_cast(unsigned short, h);
}

// A-fragment (16x32 bf16): lane L holds row L%16.
// lanes 0-15: K = {k0..k0+7, k0+16..k0+23}; lanes 16-31: +8.
__device__ __forceinline__ v16bf afrag_bf16(const unsigned short* rowptr, int k0, int lane) {
  const int klo = k0 + ((lane & 16) ? 8 : 0);
  union { u32x4 q[2]; v16bf v; } r;
  r.q[0] = *(const u32x4*)(rowptr + klo);
  r.q[1] = *(const u32x4*)(rowptr + klo + 16);
  return r.v;
}

// A-fragment built from f32 memory (convert to bf16 in registers)
__device__ __forceinline__ v16bf afrag_from_f32(const float* rowptr, int k0, int lane) {
  const int klo = k0 + ((lane & 16) ? 8 : 0);
  union { unsigned short u[16]; v16bf v; } r;
#pragma unroll
  for (int t = 0; t < 8; ++t) r.u[t]     = f2bf(rowptr[klo + t]);
#pragma unroll
  for (int t = 0; t < 8; ++t) r.u[8 + t] = f2bf(rowptr[klo + 16 + t]);
  return r.v;
}

// B-fragment (32x16 bf16): lane L holds column L%16;
// lanes 0-15: K = {k0..k0+7, k0+16..k0+23}; lanes 16-31: +8.
__device__ __forceinline__ v16bf bfrag_bf16(const unsigned short* colptr, int k0, int lane) {
  const int kb = k0 + ((lane & 16) ? 8 : 0);
  union { u32x4 q[2]; v16bf v; } r;
  r.q[0] = *(const u32x4*)(colptr + kb);
  r.q[1] = *(const u32x4*)(colptr + kb + 16);
  return r.v;
}

__device__ __forceinline__ v8f wmma_bf16(v16bf a, v16bf b, v8f c) {
  return __builtin_amdgcn_wmma_f32_16x16x32_bf16(false, a, false, b, (short)0, c,
                                                 false, false);
}

// Double-buffered 16x64 GEMM core: acc[4] += Arow(16x1024) * W(cols: 4x16)
template<bool AF32>
__device__ __forceinline__ void gemm_core(const void* arow_v,
                                          const unsigned short* wbase,
                                          int lane, v8f acc[4]) {
  const unsigned short* arow_h = (const unsigned short*)arow_v;
  const float*          arow_f = (const float*)arow_v;

  v16bf a_cur = AF32 ? afrag_from_f32(arow_f, 0, lane) : afrag_bf16(arow_h, 0, lane);
  v16bf b_cur[4];
#pragma unroll
  for (int j = 0; j < 4; ++j)
    b_cur[j] = bfrag_bf16(wbase + (size_t)(j * 16) * DMODEL, 0, lane);

  for (int k0 = 0; k0 < DMODEL; k0 += 32) {
    const int kn = (k0 + 32 < DMODEL) ? (k0 + 32) : 0;   // wrap: always in-bounds
    v16bf a_nxt = AF32 ? afrag_from_f32(arow_f, kn, lane) : afrag_bf16(arow_h, kn, lane);
    v16bf b_nxt[4];
#pragma unroll
    for (int j = 0; j < 4; ++j)
      b_nxt[j] = bfrag_bf16(wbase + (size_t)(j * 16) * DMODEL, kn, lane);

#pragma unroll
    for (int j = 0; j < 4; ++j)
      acc[j] = wmma_bf16(a_cur, b_cur[j], acc[j]);

    a_cur = a_nxt;
#pragma unroll
    for (int j = 0; j < 4; ++j) b_cur[j] = b_nxt[j];
  }
}

// ---------- kernel: weight transpose + f32->bf16 ----------
__global__ __launch_bounds__(256) void k_transpose_w(const float* __restrict__ W,
                                                     unsigned short* __restrict__ WT) {
  int idx = blockIdx.x * 256 + threadIdx.x;    // 0 .. 1M-1
  int i = idx >> 10;
  int o = idx & 1023;
  WT[(size_t)o * DMODEL + i] = f2bf(W[idx]);
}

// ---------- kernel: elementwise f32 -> bf16 (8 elems/thread) ----------
__global__ __launch_bounds__(256) void k_convert_bf16(const float* __restrict__ in,
                                                      unsigned short* __restrict__ out) {
  int idx = (blockIdx.x * 256 + threadIdx.x) * 8;
  float4 f0 = *(const float4*)(in + idx);
  float4 f1 = *(const float4*)(in + idx + 4);
  union { unsigned short u[8]; u32x4 q; } r;
  r.u[0] = f2bf(f0.x); r.u[1] = f2bf(f0.y); r.u[2] = f2bf(f0.z); r.u[3] = f2bf(f0.w);
  r.u[4] = f2bf(f1.x); r.u[5] = f2bf(f1.y); r.u[6] = f2bf(f1.z); r.u[7] = f2bf(f1.w);
  *(u32x4*)(out + idx) = r.q;
}

// ---------- kernel: Q/K/V projection ----------
// One wave computes a 16-row x 64-col tile (one head). Output bf16 [b][h][n][64],
// scaled by outScale (1/sqrt(d) for Q, 1.0 for K/V).
template<int RPB, bool AF32>
__global__ __launch_bounds__(256) void k_gemm_proj(const void* __restrict__ A,
                                                   const unsigned short* __restrict__ WT,
                                                   unsigned short* __restrict__ Out,
                                                   float outScale) {
  const int lane = threadIdx.x & 31;
  const int wave = threadIdx.x >> 5;
  const int tile = blockIdx.x * 8 + wave;
  const int row0 = (tile >> 4) << 4;
  const int h    = tile & 15;

  v8f acc[4];
#pragma unroll
  for (int j = 0; j < 4; ++j)
#pragma unroll
    for (int e = 0; e < 8; ++e) acc[j][e] = 0.f;

  const void* arow = AF32
      ? (const void*)((const float*)A + (size_t)(row0 + (lane & 15)) * DMODEL)
      : (const void*)((const unsigned short*)A + (size_t)(row0 + (lane & 15)) * DMODEL);
  const unsigned short* wbase = WT + (size_t)(h * DIM_HEAD + (lane & 15)) * DMODEL;

  gemm_core<AF32>(arow, wbase, lane, acc);

  const int mbase = (lane & 16) ? 8 : 0;
#pragma unroll
  for (int j = 0; j < 4; ++j) {
    const int d = j * 16 + (lane & 15);
#pragma unroll
    for (int r = 0; r < 8; ++r) {
      int row = row0 + mbase + r;
      int b   = row / RPB;              // compile-time power of 2 -> shift
      int nr  = row & (RPB - 1);
      Out[(((size_t)b * HEADS + h) * RPB + nr) * DIM_HEAD + d] = f2bf(acc[j][r] * outScale);
    }
  }
}

// ---------- kernel: flash attention (fully transposed score pipeline) ----------
// S^T = K(keys x d) @ Q^T(d x queries): each lane owns ONE query (lane%16) and
// 16 of 32 keys in-register -> softmax = in-lane reduce + one shfl_xor(16).
// O^T = V^T(d x keys) @ P^T(keys x queries): alpha & 1/l are per-lane scalars.
__global__ __launch_bounds__(256) void k_attn(const unsigned short* __restrict__ Qb,
                                              const unsigned short* __restrict__ Kb,
                                              const unsigned short* __restrict__ Vb,
                                              unsigned short* __restrict__ Ob) {
  __shared__ unsigned short Vt[DIM_HEAD * 32];   // V^T chunk: [d][key]
  __shared__ unsigned short Pl[8][16 * 32];      // per-wave P tile [q][key]

  const int lane = threadIdx.x & 31;
  const int wave = threadIdx.x >> 5;
  const int q    = lane & 15;                    // this lane's query (in wave tile)
  const int half = (lane >> 4) & 1;
  const int bh   = blockIdx.x >> 4;              // b*16 + h
  const int qc   = blockIdx.x & 15;
  const int q0   = qc * 128 + wave * 16;

  const unsigned short* Kbase = Kb + (size_t)bh * NK * DIM_HEAD;
  const unsigned short* Vbase = Vb + (size_t)bh * NK * DIM_HEAD;

  // Q^T B-fragments, register-resident (lane = query column, K = d contiguous)
  const unsigned short* Qrow = Qb + ((size_t)bh * NQ + q0 + q) * DIM_HEAD;
  v16bf qfB[2];
  qfB[0] = bfrag_bf16(Qrow, 0,  lane);
  qfB[1] = bfrag_bf16(Qrow, 32, lane);

  v8f o[4];                                      // O^T tiles: rows d=16j.., col q
#pragma unroll
  for (int j = 0; j < 4; ++j)
#pragma unroll
    for (int e = 0; e < 8; ++e) o[j][e] = 0.f;
  float m_l = -3.0e38f, l_l = 0.f;               // per-lane (per-query) stats

  for (int m0 = 0; m0 < NK; m0 += 32) {
    if (m0 + 32 < NK) {                          // global_prefetch_b8 next chunk
      __builtin_prefetch(Kbase + (size_t)(m0 + 32 + lane) * DIM_HEAD, 0, 0);
      __builtin_prefetch(Vbase + (size_t)(m0 + 32 + lane) * DIM_HEAD, 0, 0);
    }

    // cooperatively stage V^T chunk (32 keys x 64 d, transposed)
#pragma unroll
    for (int e = 0; e < 8; ++e) {
      int idx = threadIdx.x + 256 * e;           // 0..2047
      int kl = idx >> 6, d = idx & 63;
      Vt[d * 32 + kl] = Vbase[(size_t)(m0 + kl) * DIM_HEAD + d];
    }
    __syncthreads();

    // S^T tiles: st[s] covers keys m0+16s .. +15 (rows) x 16 queries (cols)
    v8f st[2];
#pragma unroll
    for (int s = 0; s < 2; ++s) {
      const unsigned short* krow = Kbase + (size_t)(m0 + 16 * s + q) * DIM_HEAD;
      v16bf ka0 = afrag_bf16(krow, 0,  lane);
      v16bf ka1 = afrag_bf16(krow, 32, lane);
      v8f z;
#pragma unroll
      for (int e = 0; e < 8; ++e) z[e] = 0.f;
      st[s] = wmma_bf16(ka0, qfB[0], z);
      st[s] = wmma_bf16(ka1, qfB[1], st[s]);
    }

    // per-lane online softmax over this lane's 16 keys + one cross-half combine
    float mx = st[0][0];
#pragma unroll
    for (int r = 1; r < 8; ++r) mx = fmaxf(mx, st[0][r]);
#pragma unroll
    for (int r = 0; r < 8; ++r) mx = fmaxf(mx, st[1][r]);
    mx = fmaxf(mx, __shfl_xor(mx, 16, 32));
    float mnew  = fmaxf(m_l, mx);
    float alpha = __expf(m_l - mnew);

    union { unsigned short u[8]; u32x4 v; } pk0, pk1;
    float rs = 0.f;
#pragma unroll
    for (int r = 0; r < 8; ++r) {
      float p = __expf(st[0][r] - mnew);
      rs += p; pk0.u[r] = f2bf(p);
    }
#pragma unroll
    for (int r = 0; r < 8; ++r) {
      float p = __expf(st[1][r] - mnew);
      rs += p; pk1.u[r] = f2bf(p);
    }
    rs += __shfl_xor(rs, 16, 32);
    l_l = l_l * alpha + rs;
    m_l = mnew;
#pragma unroll
    for (int j = 0; j < 4; ++j) o[j] *= alpha;

    // P^T -> LDS, packed b128 stores: Pl[q][key], key = 16s + 8*half + r
    unsigned short* pl = &Pl[wave][0];
    *(u32x4*)(pl + q * 32 + 8 * half)      = pk0.v;
    *(u32x4*)(pl + q * 32 + 16 + 8 * half) = pk1.v;

    // O^T += V^T @ P^T  (B-frag shared across the 4 d-tiles)
    v16bf pb = bfrag_bf16(pl + q * 32, 0, lane);
#pragma unroll
    for (int j = 0; j < 4; ++j) {
      v16bf va = afrag_bf16(Vt + (size_t)(16 * j + q) * 32, 0, lane);
      o[j] = wmma_bf16(va, pb, o[j]);
    }
    __syncthreads();
  }

  // epilogue: O^T/l -> Ob[b][n][h*64+d], packed b128 stores (d contiguous)
  const int b = bh >> 4, h = bh & 15;
  const float inv = 1.0f / l_l;
  unsigned short* orow = Ob + ((size_t)(b * NQ + q0 + q)) * DMODEL + h * DIM_HEAD;
#pragma unroll
  for (int j = 0; j < 4; ++j) {
    union { unsigned short u[8]; u32x4 v; } pk;
#pragma unroll
    for (int r = 0; r < 8; ++r) pk.u[r] = f2bf(o[j][r] * inv);
    *(u32x4*)(orow + 16 * j + 8 * half) = pk.v;
  }
}

// ---------- kernel: output projection + bias (f32 out) ----------
__global__ __launch_bounds__(256) void k_gemm_out(const unsigned short* __restrict__ A,
                                                  const unsigned short* __restrict__ WT,
                                                  const float* __restrict__ bias,
                                                  float* __restrict__ Out) {
  const int lane = threadIdx.x & 31;
  const int wave = threadIdx.x >> 5;
  const int tile = blockIdx.x * 8 + wave;
  const int row0 = (tile >> 4) << 4;
  const int ct   = tile & 15;

  v8f acc[4];
#pragma unroll
  for (int j = 0; j < 4; ++j)
#pragma unroll
    for (int e = 0; e < 8; ++e) acc[j][e] = 0.f;

  const unsigned short* arow  = A  + (size_t)(row0 + (lane & 15)) * DMODEL;
  const unsigned short* wbase = WT + (size_t)(ct * 64 + (lane & 15)) * DMODEL;

  gemm_core<false>(arow, wbase, lane, acc);

  const int mbase = (lane & 16) ? 8 : 0;
#pragma unroll
  for (int j = 0; j < 4; ++j) {
    int col = ct * 64 + j * 16 + (lane & 15);
    float bv = bias[col];
#pragma unroll
    for (int r = 0; r < 8; ++r)
      Out[(size_t)(row0 + mbase + r) * DMODEL + col] = acc[j][r] + bv;
  }
}

// ---------- launcher ----------
extern "C" void kernel_launch(void* const* d_in, const int* in_sizes, int n_in,
                              void* d_out, int out_size, void* d_ws, size_t ws_size,
                              hipStream_t stream) {
  const float* x   = (const float*)d_in[0];
  const float* ctx = (const float*)d_in[1];
  const float* Wq  = (const float*)d_in[2];
  const float* Wk  = (const float*)d_in[3];
  const float* Wv  = (const float*)d_in[4];
  const float* Wo  = (const float*)d_in[5];
  const float* bo  = (const float*)d_in[6];
  float* out = (float*)d_out;

  const size_t MB = 1024 * 1024;
  char* ws = (char*)d_ws;
  unsigned short* WqT = (unsigned short*)(ws + 0 * MB);
  unsigned short* WkT = (unsigned short*)(ws + 2 * MB);
  unsigned short* WvT = (unsigned short*)(ws + 4 * MB);
  unsigned short* WoT = (unsigned short*)(ws + 6 * MB);

  const bool big = ws_size >= 80 * MB;   // deterministic path selection
  if (ws_size < 56 * MB) return;

  // weights -> bf16 transposed [out][in]
  k_transpose_w<<<4096, 256, 0, stream>>>(Wq, WqT);
  k_transpose_w<<<4096, 256, 0, stream>>>(Wk, WkT);
  k_transpose_w<<<4096, 256, 0, stream>>>(Wv, WvT);
  k_transpose_w<<<4096, 256, 0, stream>>>(Wo, WoT);

  if (big) {
    unsigned short* xb   = (unsigned short*)(ws + 8 * MB);   // 16MB
    unsigned short* cb   = (unsigned short*)(ws + 24 * MB);  // 8MB
    unsigned short* Qb   = (unsigned short*)(ws + 32 * MB);  // 16MB
    unsigned short* Kb   = (unsigned short*)(ws + 48 * MB);  // 8MB
    unsigned short* Vb   = (unsigned short*)(ws + 56 * MB);  // 8MB
    unsigned short* Obuf = (unsigned short*)(ws + 64 * MB);  // 16MB

    k_convert_bf16<<<4096, 256, 0, stream>>>(x,   xb);
    k_convert_bf16<<<2048, 256, 0, stream>>>(ctx, cb);

    k_gemm_proj<NQ, false><<<1024, 256, 0, stream>>>(xb, WqT, Qb, ATTN_SCALE);
    k_gemm_proj<NK, false><<<512,  256, 0, stream>>>(cb, WkT, Kb, 1.0f);
    k_gemm_proj<NK, false><<<512,  256, 0, stream>>>(cb, WvT, Vb, 1.0f);

    k_attn<<<1024, 256, 0, stream>>>(Qb, Kb, Vb, Obuf);
    k_gemm_out<<<1024, 256, 0, stream>>>(Obuf, WoT, bo, out);
  } else {
    unsigned short* Qb   = (unsigned short*)(ws + 8 * MB);   // 16MB
    unsigned short* Kb   = (unsigned short*)(ws + 24 * MB);  // 8MB
    unsigned short* Vb   = (unsigned short*)(ws + 32 * MB);  // 8MB
    unsigned short* Obuf = (unsigned short*)(ws + 40 * MB);  // 16MB

    k_gemm_proj<NQ, true><<<1024, 256, 0, stream>>>(x,   WqT, Qb, ATTN_SCALE);
    k_gemm_proj<NK, true><<<512,  256, 0, stream>>>(ctx, WkT, Kb, 1.0f);
    k_gemm_proj<NK, true><<<512,  256, 0, stream>>>(ctx, WvT, Vb, 1.0f);

    k_attn<<<1024, 256, 0, stream>>>(Qb, Kb, Vb, Obuf);
    k_gemm_out<<<1024, 256, 0, stream>>>(Obuf, WoT, bo, out);
  }
}
